// SelfAttention_49108656062912
// MI455X (gfx1250) — compile-verified, matrix operand order checked by reading, not run on hardware
//
#include <hip/hip_runtime.h>

// ---------------------------------------------------------------------------
// CDNA5 (gfx1250) edge-conditioned multi-head graph attention.
// All 128-deep contractions on v_wmma_f32_16x16x32_bf16 (f32 accum).
// Weights pre-packed to WMMA B-operand order (bf16); activations staged in
// LDS in WMMA A-operand order so operand loads are single 32B vector reads.
// ---------------------------------------------------------------------------

typedef __bf16 bf16;
typedef bf16  v16bf __attribute__((ext_vector_type(16)));
typedef float v8f   __attribute__((ext_vector_type(8)));

#define TE 64          // edges per block in edge kernels
#define NT 256         // threads per block (8 waves)

__device__ inline v8f wmma_bf16(v16bf a, v16bf b, v8f c) {
  return __builtin_amdgcn_wmma_f32_16x16x32_bf16(false, a, false, b, (short)0, c,
                                                 false, false);
}

// ---- monotone float<->uint encoding for atomic segment-max -----------------
__device__ inline unsigned fenc(float f) {
  unsigned u = __float_as_uint(f);
  return (u & 0x80000000u) ? ~u : (u | 0x80000000u);
}
__device__ inline float fdec(unsigned e) {
  return (e & 0x80000000u) ? __uint_as_float(e & 0x7FFFFFFFu) : __uint_as_float(~e);
}

// ---------------------------------------------------------------------------
__global__ __launch_bounds__(NT)
void zero_kernel(float* p, int n) {
  int i = blockIdx.x * NT + threadIdx.x;
  if (i < n) p[i] = 0.0f;
}

// Pack W[128 x ncols] (row-major, K-major) into WMMA B-operand order, bf16.
// Slot s = (tn*4+kk)*32+lane holds 16 halves = one lane's v16bf operand.
// Zero-pads columns >= ncols (used for the 128x8 ea_w2).
__global__ __launch_bounds__(NT)
void pack_w_kernel(const float* __restrict__ W, bf16* __restrict__ P,
                   int ncols, int ntn) {
  int s = blockIdx.x * NT + threadIdx.x;
  if (s >= ntn * 128) return;
  int lane = s & 31, kk = (s >> 5) & 3, tn = s >> 7;
  int n  = tn * 16 + (lane & 15);
  int k0 = kk * 32 + ((lane >> 4) & 1) * 8;
  v16bf b;
#pragma unroll
  for (int j = 0; j < 8; ++j) {
    b[j]     = (n < ncols) ? (bf16)W[(size_t)(k0 + j) * ncols + n]      : (bf16)0.0f;
    b[8 + j] = (n < ncols) ? (bf16)W[(size_t)(k0 + 16 + j) * ncols + n] : (bf16)0.0f;
  }
  *(v16bf*)(P + (size_t)s * 16) = b;
}

// ---------------------------------------------------------------------------
// C[N x 128] = bf16(A) @ Wp + bias.  128x128 per block, A staged swizzled in
// LDS, 32 WMMAs per wave, B tile = one coalesced 32B load per lane.
__global__ __launch_bounds__(NT)
void gemm128_kernel(const float* __restrict__ A, const bf16* __restrict__ Wp,
                    const float* __restrict__ bias, float* __restrict__ C,
                    int nrows) {
  __shared__ __align__(32) bf16 As[8 * 4 * 32 * 16];   // 32 KB, swizzled
  int tid  = threadIdx.x;
  int row0 = blockIdx.x * 128;
  if (row0 >= nrows) return;

  for (int s = tid; s < 1024; s += NT) {
    int lane = s & 31, kk = (s >> 5) & 3, tm = s >> 7;
    int m = lane & 15, kh = (lane >> 4) & 1;
    const float* p = A + (size_t)(row0 + tm * 16 + m) * 128 + kk * 32 + kh * 8;
    v16bf a;
#pragma unroll
    for (int j = 0; j < 8; ++j) { a[j] = (bf16)p[j]; a[8 + j] = (bf16)p[16 + j]; }
    *(v16bf*)(As + s * 16) = a;
  }
  __syncthreads();

  int wave = tid >> 5, lane = tid & 31;
  v8f acc[8] = {};
#pragma unroll
  for (int kk = 0; kk < 4; ++kk) {
    v16bf b = *(const v16bf*)(Wp + (size_t)((wave * 4 + kk) * 32 + lane) * 16);
#pragma unroll
    for (int tm = 0; tm < 8; ++tm) {
      v16bf a = *(const v16bf*)(As + ((tm * 4 + kk) * 32 + lane) * 16);
      acc[tm] = wmma_bf16(a, b, acc[tm]);
    }
  }
  int n = lane & 15, mo = (lane >> 4) * 8;
  float bb = bias[wave * 16 + n];
#pragma unroll
  for (int tm = 0; tm < 8; ++tm)
#pragma unroll
    for (int r = 0; r < 8; ++r)
      C[(size_t)(row0 + tm * 16 + mo + r) * 128 + wave * 16 + n] = acc[tm][r] + bb;
}

// ---------------------------------------------------------------------------
// shared helpers for edge kernels
__device__ inline void mlp_first_layer(const float* __restrict__ ef_s,
                                       const float* __restrict__ w1,
                                       const float* __restrict__ b1,
                                       float* __restrict__ hbuf, int tid) {
  for (int idx = tid; idx < TE * 128; idx += NT) {
    int e = idx >> 7, d = idx & 127;
    float acc = b1[d];
#pragma unroll
    for (int f = 0; f < 5; ++f) acc += ef_s[e * 5 + f] * w1[f * 128 + d];
    hbuf[idx] = fmaxf(acc, 0.0f);
  }
}

// 4 threads/edge partial stats + quad shfl reduction
__device__ inline void ln_stats(const float* __restrict__ hbuf,
                                float* __restrict__ mu_s, float* __restrict__ rs_s,
                                int tid) {
  int e = tid >> 2, p = tid & 3;
  float s = 0.f, s2 = 0.f;
#pragma unroll 4
  for (int j = 0; j < 32; ++j) {
    float x = hbuf[e * 128 + p * 32 + j];
    s += x; s2 += x * x;
  }
  s  += __shfl_xor(s, 1);  s2 += __shfl_xor(s2, 1);
  s  += __shfl_xor(s, 2);  s2 += __shfl_xor(s2, 2);
  if (p == 0) {
    float mu  = s * (1.0f / 128.0f);
    float var = s2 * (1.0f / 128.0f) - mu * mu;
    mu_s[e] = mu;
    rs_s[e] = rsqrtf(var + 1e-5f);
  }
}

// normalize + write bf16 staging buffer directly in WMMA A-operand order
__device__ inline void norm_swizzle(const float* __restrict__ hbuf,
                                    bf16* __restrict__ hk,
                                    const float* __restrict__ g,
                                    const float* __restrict__ beta,
                                    const float* __restrict__ mu_s,
                                    const float* __restrict__ rs_s, int tid) {
  for (int s = tid; s < (TE / 16) * 128; s += NT) {      // 512 slots
    int lane = s & 31, kk = (s >> 5) & 3, tm = s >> 7;
    int e  = tm * 16 + (lane & 15);
    int d0 = kk * 32 + ((lane >> 4) & 1) * 8;
    float mu = mu_s[e], rs = rs_s[e];
    v16bf a;
#pragma unroll
    for (int j = 0; j < 8; ++j) {
      int d  = d0 + j;
      int d2 = d0 + 16 + j;
      a[j]     = (bf16)((hbuf[e * 128 + d]  - mu) * rs * g[d]  + beta[d]);
      a[8 + j] = (bf16)((hbuf[e * 128 + d2] - mu) * rs * g[d2] + beta[d2]);
    }
    *(v16bf*)(hk + s * 16) = a;
  }
}

// ---------------------------------------------------------------------------
// Per-edge logits: w[e][h] = <q[head], k[tail] + ekMLP(ef)>_h + eaMLP(ef)[h]
__global__ __launch_bounds__(NT)
void edge_logits_kernel(const float* __restrict__ ef,
                        const int* __restrict__ bidx, const int* __restrict__ hidx,
                        const int* __restrict__ tidx,
                        const float* __restrict__ ekw1, const float* __restrict__ ekb1,
                        const float* __restrict__ ekg,  const float* __restrict__ ekbeta,
                        const bf16*  __restrict__ ekw2p, const float* __restrict__ ekb2,
                        const float* __restrict__ eaw1, const float* __restrict__ eab1,
                        const float* __restrict__ eag,  const float* __restrict__ eabeta,
                        const bf16*  __restrict__ eaw2p, const float* __restrict__ eab2,
                        const float* __restrict__ qbuf, const float* __restrict__ kbuf,
                        float* __restrict__ wlog, int E, int L) {
  __shared__ float hbuf[TE * 128];                 // pre-LN acts / ek output (f32)
  __shared__ __align__(32) bf16 hk[(TE / 16) * 128 * 16];  // 16 KB swizzled bf16
  __shared__ float mu_s[TE], rs_s[TE];
  __shared__ float ea_s[TE * 8];
  __shared__ float ef_s[TE * 5];
  __shared__ int   eH[TE], eT[TE];

  int tid  = threadIdx.x;
  int e0   = blockIdx.x * TE;
  int wave = tid >> 5, lane = tid & 31;

  for (int i = tid; i < TE; i += NT) {
    int eg = e0 + i;
    if (eg < E) { eH[i] = bidx[eg] * L + hidx[eg]; eT[i] = bidx[eg] * L + tidx[eg]; }
    else        { eH[i] = 0; eT[i] = 0; }
  }
  for (int i = tid; i < TE * 5; i += NT) {
    int eg = e0 + i / 5;
    ef_s[i] = (eg < E) ? ef[(size_t)eg * 5 + (i % 5)] : 0.0f;
  }
  __syncthreads();

  // ---- ea MLP first layer + LN --------------------------------------------
  mlp_first_layer(ef_s, eaw1, eab1, hbuf, tid);
  __syncthreads();
  ln_stats(hbuf, mu_s, rs_s, tid);
  __syncthreads();
  norm_swizzle(hbuf, hk, eag, eabeta, mu_s, rs_s, tid);
  __syncthreads();

  // ---- ea second layer via WMMA (128x16, cols 8..15 zero-padded);
  //      overlapped with ek first layer on the remaining waves --------------
  if (wave < 4) {
    int tm = wave;
    v8f acc = {};
#pragma unroll
    for (int kk = 0; kk < 4; ++kk) {
      v16bf b = *(const v16bf*)(eaw2p + (size_t)(kk * 32 + lane) * 16);
      v16bf a = *(const v16bf*)(hk + ((tm * 4 + kk) * 32 + lane) * 16);
      acc = wmma_bf16(a, b, acc);
    }
    int n = lane & 15, mo = (lane >> 4) * 8;
    if (n < 8) {
#pragma unroll
      for (int r = 0; r < 8; ++r)
        ea_s[(tm * 16 + mo + r) * 8 + n] = acc[r] + eab2[n];
    }
  }
  // ---- ek MLP first layer (all threads) -----------------------------------
  mlp_first_layer(ef_s, ekw1, ekb1, hbuf, tid);
  __syncthreads();
  ln_stats(hbuf, mu_s, rs_s, tid);
  __syncthreads();
  norm_swizzle(hbuf, hk, ekg, ekbeta, mu_s, rs_s, tid);
  __syncthreads();

  // ---- ek second layer: (64 x 128) @ (128 x 128) via WMMA -----------------
  {
    int tn = wave;
    v8f acc[4] = {};
#pragma unroll
    for (int kk = 0; kk < 4; ++kk) {
      v16bf b = *(const v16bf*)(ekw2p + (size_t)((tn * 4 + kk) * 32 + lane) * 16);
#pragma unroll
      for (int tm = 0; tm < 4; ++tm) {
        v16bf a = *(const v16bf*)(hk + ((tm * 4 + kk) * 32 + lane) * 16);
        acc[tm] = wmma_bf16(a, b, acc[tm]);
      }
    }
    int n = lane & 15, mo = (lane >> 4) * 8;
#pragma unroll
    for (int tm = 0; tm < 4; ++tm)
#pragma unroll
      for (int r = 0; r < 8; ++r)
        hbuf[(tm * 16 + mo + r) * 128 + tn * 16 + n] = acc[tm][r];
  }
  __syncthreads();

  // ---- logits: per-head 16-dim dot with gathered q/k (float4 loads) -------
  {
    int e = tid >> 2, p = tid & 3;
    int eg = e0 + e;
    if (eg < E) {
      const float4* q4 = (const float4*)(qbuf + (size_t)eH[e] * 128 + p * 32);
      const float4* k4 = (const float4*)(kbuf + (size_t)eT[e] * 128 + p * 32);
      const float4* b4 = (const float4*)(ekb2 + p * 32);
      float s0 = 0.f, s1 = 0.f;
#pragma unroll
      for (int jj = 0; jj < 8; ++jj) {
        float4 qv = q4[jj], kv = k4[jj], bv = b4[jj];
        int d = p * 32 + jj * 4;
        float t = (kv.x + hbuf[e * 128 + d + 0] + bv.x) * qv.x
                + (kv.y + hbuf[e * 128 + d + 1] + bv.y) * qv.y
                + (kv.z + hbuf[e * 128 + d + 2] + bv.z) * qv.z
                + (kv.w + hbuf[e * 128 + d + 3] + bv.w) * qv.w;
        if (jj < 4) s0 += t; else s1 += t;
      }
      wlog[(size_t)eg * 8 + 2 * p]     = s0 + ea_s[e * 8 + 2 * p];
      wlog[(size_t)eg * 8 + 2 * p + 1] = s1 + ea_s[e * 8 + 2 * p + 1];
    }
  }
}

// ---------------------------------------------------------------------------
__global__ __launch_bounds__(NT)
void seg_max_kernel(const float* __restrict__ wlog, const int* __restrict__ bidx,
                    const int* __restrict__ hidx, unsigned* __restrict__ mbits,
                    int E, int L) {
  int i = blockIdx.x * NT + threadIdx.x;
  if (i >= E * 8) return;
  int e = i >> 3, h = i & 7;
  int node = bidx[e] * L + hidx[e];
  atomicMax(&mbits[node * 8 + h], fenc(wlog[i]));
}

__global__ __launch_bounds__(NT)
void seg_exp_kernel(float* __restrict__ wlog, const int* __restrict__ bidx,
                    const int* __restrict__ hidx, const unsigned* __restrict__ mbits,
                    float* __restrict__ denom, int E, int L) {
  int i = blockIdx.x * NT + threadIdx.x;
  if (i >= E * 8) return;
  int e = i >> 3, h = i & 7;
  int node = bidx[e] * L + hidx[e];
  float ex = expf(wlog[i] - fdec(mbits[node * 8 + h]));
  wlog[i] = ex;
  atomicAdd(&denom[node * 8 + h], ex);
}

// ---------------------------------------------------------------------------
// Combine: recompute ev MLP (WMMA), y[head] += att * (v[tail] + ev)
__global__ __launch_bounds__(NT)
void edge_combine_kernel(const float* __restrict__ ef,
                         const int* __restrict__ bidx, const int* __restrict__ hidx,
                         const int* __restrict__ tidx,
                         const float* __restrict__ evw1, const float* __restrict__ evb1,
                         const float* __restrict__ evg,  const float* __restrict__ evbeta,
                         const bf16*  __restrict__ evw2p, const float* __restrict__ evb2,
                         const float* __restrict__ vbuf, const float* __restrict__ exbuf,
                         const float* __restrict__ denom, float* __restrict__ ybuf,
                         int E, int L) {
  __shared__ float hbuf[TE * 128];
  __shared__ __align__(32) bf16 hv[(TE / 16) * 128 * 16];
  __shared__ float mu_s[TE], rs_s[TE];
  __shared__ float ef_s[TE * 5];
  __shared__ int   eH[TE], eT[TE];

  int tid  = threadIdx.x;
  int e0   = blockIdx.x * TE;
  int wave = tid >> 5, lane = tid & 31;

  for (int i = tid; i < TE; i += NT) {
    int eg = e0 + i;
    if (eg < E) { eH[i] = bidx[eg] * L + hidx[eg]; eT[i] = bidx[eg] * L + tidx[eg]; }
    else        { eH[i] = 0; eT[i] = 0; }
  }
  for (int i = tid; i < TE * 5; i += NT) {
    int eg = e0 + i / 5;
    ef_s[i] = (eg < E) ? ef[(size_t)eg * 5 + (i % 5)] : 0.0f;
  }
  __syncthreads();

  mlp_first_layer(ef_s, evw1, evb1, hbuf, tid);
  __syncthreads();
  ln_stats(hbuf, mu_s, rs_s, tid);
  __syncthreads();
  norm_swizzle(hbuf, hv, evg, evbeta, mu_s, rs_s, tid);
  __syncthreads();

  {
    int tn = wave;
    v8f acc[4] = {};
#pragma unroll
    for (int kk = 0; kk < 4; ++kk) {
      v16bf b = *(const v16bf*)(evw2p + (size_t)((tn * 4 + kk) * 32 + lane) * 16);
#pragma unroll
      for (int tm = 0; tm < 4; ++tm) {
        v16bf a = *(const v16bf*)(hv + ((tm * 4 + kk) * 32 + lane) * 16);
        acc[tm] = wmma_bf16(a, b, acc[tm]);
      }
    }
    int n = lane & 15, mo = (lane >> 4) * 8;
#pragma unroll
    for (int tm = 0; tm < 4; ++tm)
#pragma unroll
      for (int r = 0; r < 8; ++r)
        hbuf[(tm * 16 + mo + r) * 128 + tn * 16 + n] = acc[tm][r];
  }
  __syncthreads();

  // weighted scatter: 4 threads/edge, 32 dims (2 heads) each
  {
    int e = tid >> 2, p = tid & 3;
    int eg = e0 + e;
    if (eg < E) {
      int node = eH[e];
      float a0 = exbuf[(size_t)eg * 8 + 2 * p]     / (denom[node * 8 + 2 * p]     + 1e-16f);
      float a1 = exbuf[(size_t)eg * 8 + 2 * p + 1] / (denom[node * 8 + 2 * p + 1] + 1e-16f);
      const float4* v4 = (const float4*)(vbuf + (size_t)eT[e] * 128 + p * 32);
      const float4* b4 = (const float4*)(evb2 + p * 32);
      float* yrow = ybuf + (size_t)node * 128 + p * 32;
#pragma unroll
      for (int jj = 0; jj < 8; ++jj) {
        float4 vv = v4[jj], bv = b4[jj];
        int d = p * 32 + jj * 4;
        float att = (jj < 4) ? a0 : a1;
        atomicAdd(&yrow[jj * 4 + 0], att * (vv.x + hbuf[e * 128 + d + 0] + bv.x));
        atomicAdd(&yrow[jj * 4 + 1], att * (vv.y + hbuf[e * 128 + d + 1] + bv.y));
        atomicAdd(&yrow[jj * 4 + 2], att * (vv.z + hbuf[e * 128 + d + 2] + bv.z));
        atomicAdd(&yrow[jj * 4 + 3], att * (vv.w + hbuf[e * 128 + d + 3] + bv.w));
      }
    }
  }
}

// ---------------------------------------------------------------------------
extern "C" void kernel_launch(void* const* d_in, const int* in_sizes, int n_in,
                              void* d_out, int out_size, void* d_ws, size_t ws_size,
                              hipStream_t stream) {
  const float* key        = (const float*)d_in[0];
  const float* value      = (const float*)d_in[1];
  const float* query      = (const float*)d_in[2];
  const float* edge_feats = (const float*)d_in[3];
  const int*   bidx = (const int*)d_in[5];
  const int*   hidx = (const int*)d_in[6];
  const int*   tidx = (const int*)d_in[7];
  const float* key_w   = (const float*)d_in[8];  const float* key_b   = (const float*)d_in[9];
  const float* query_w = (const float*)d_in[10]; const float* query_b = (const float*)d_in[11];
  const float* value_w = (const float*)d_in[12]; const float* value_b = (const float*)d_in[13];
  const float* proj_w  = (const float*)d_in[14]; const float* proj_b  = (const float*)d_in[15];
  const float* eaw1 = (const float*)d_in[16]; const float* eab1 = (const float*)d_in[17];
  const float* eag  = (const float*)d_in[18]; const float* eabt = (const float*)d_in[19];
  const float* eaw2 = (const float*)d_in[20]; const float* eab2 = (const float*)d_in[21];
  const float* ekw1 = (const float*)d_in[22]; const float* ekb1 = (const float*)d_in[23];
  const float* ekg  = (const float*)d_in[24]; const float* ekbt = (const float*)d_in[25];
  const float* ekw2 = (const float*)d_in[26]; const float* ekb2 = (const float*)d_in[27];
  const float* evw1 = (const float*)d_in[28]; const float* evb1 = (const float*)d_in[29];
  const float* evg  = (const float*)d_in[30]; const float* evbt = (const float*)d_in[31];
  const float* evw2 = (const float*)d_in[32]; const float* evb2 = (const float*)d_in[33];

  const int N = in_sizes[0] / 128;   // B*L = 16384
  const int E = in_sizes[5];         // 262144
  const int L = 4096;

  // ws layout (floats): q | k | v | wlog(E*8) | mbits(N*8) | denom(N*8) | y | packed weights
  float* ws    = (float*)d_ws;
  float* qbuf  = ws;
  float* kbuf  = qbuf + (size_t)N * 128;
  float* vbuf  = kbuf + (size_t)N * 128;
  float* wlog  = vbuf + (size_t)N * 128;
  unsigned* mbits = (unsigned*)(wlog + (size_t)E * 8);
  float* denom = (float*)(mbits + (size_t)N * 8);
  float* ybuf  = denom + (size_t)N * 8;
  bf16* pk = (bf16*)(ybuf + (size_t)N * 128);   // packed weights region
  bf16* qw_p = pk;                    // each full matrix: 8*128*16 = 16384 halves
  bf16* kw_p = qw_p + 16384;
  bf16* vw_p = kw_p + 16384;
  bf16* pw_p = vw_p + 16384;
  bf16* ekw2_p = pw_p + 16384;
  bf16* evw2_p = ekw2_p + 16384;
  bf16* eaw2_p = evw2_p + 16384;      // 1 tile-col: 128*16 = 2048 halves

  // zero mbits | denom | y (contiguous)
  int zn = N * 8 * 2 + N * 128;
  zero_kernel<<<(zn + NT - 1) / NT, NT, 0, stream>>>((float*)mbits, zn);

  // pack weights to WMMA operand order (bf16)
  pack_w_kernel<<<4, NT, 0, stream>>>(query_w, qw_p, 128, 8);
  pack_w_kernel<<<4, NT, 0, stream>>>(key_w,   kw_p, 128, 8);
  pack_w_kernel<<<4, NT, 0, stream>>>(value_w, vw_p, 128, 8);
  pack_w_kernel<<<4, NT, 0, stream>>>(proj_w,  pw_p, 128, 8);
  pack_w_kernel<<<4, NT, 0, stream>>>(ekw2,    ekw2_p, 128, 8);
  pack_w_kernel<<<4, NT, 0, stream>>>(evw2,    evw2_p, 128, 8);
  pack_w_kernel<<<1, NT, 0, stream>>>(eaw2,    eaw2_p, 8,   1);

  // node projections (q/k/v)
  gemm128_kernel<<<(N + 127) / 128, NT, 0, stream>>>(query, qw_p, query_b, qbuf, N);
  gemm128_kernel<<<(N + 127) / 128, NT, 0, stream>>>(key,   kw_p, key_b,   kbuf, N);
  gemm128_kernel<<<(N + 127) / 128, NT, 0, stream>>>(value, vw_p, value_b, vbuf, N);

  int eb = (E + TE - 1) / TE;
  edge_logits_kernel<<<eb, NT, 0, stream>>>(edge_feats, bidx, hidx, tidx,
                                            ekw1, ekb1, ekg, ekbt, ekw2_p, ekb2,
                                            eaw1, eab1, eag, eabt, eaw2_p, eab2,
                                            qbuf, kbuf, wlog, E, L);

  int sb = (E * 8 + NT - 1) / NT;
  seg_max_kernel<<<sb, NT, 0, stream>>>(wlog, bidx, hidx, mbits, E, L);
  seg_exp_kernel<<<sb, NT, 0, stream>>>(wlog, bidx, hidx, mbits, denom, E, L);

  edge_combine_kernel<<<eb, NT, 0, stream>>>(edge_feats, bidx, hidx, tidx,
                                             evw1, evb1, evg, evbt, evw2_p, evb2,
                                             vbuf, wlog, denom, ybuf, E, L);

  // output projection -> d_out
  gemm128_kernel<<<(N + 127) / 128, NT, 0, stream>>>(ybuf, pw_p, proj_b,
                                                     (float*)d_out, N);
}
